// STBlock_19275813224643
// MI455X (gfx1250) — compile-verified, matrix operand order checked by reading, not run on hardware
//
#include <hip/hip_runtime.h>

// ---------- problem constants ----------
#define Bc 16
#define Cc 64
#define Nc 512
#define Tc 64
#define Kc 3
#define NE ((size_t)Bc * Cc * Nc * Tc)   // 33,554,432 elements per tensor
#define CNT ((float)(Bc * Nc * Tc))      // BN reduction count = 524288
#define EPS 1e-5f

typedef __bf16 v16bf __attribute__((ext_vector_type(16)));
typedef float  v8f   __attribute__((ext_vector_type(8)));

__device__ __forceinline__ v8f wmma_bf16(v16bf a, v16bf b, v8f c) {
  // D = A(16x32) * B(32x16) + C, f32 accumulate
  return __builtin_amdgcn_wmma_f32_16x16x32_bf16(false, a, false, b, (short)0, c,
                                                 false, false);
}

// ---- CDNA5 async global->LDS copy (ASYNCcnt-tracked, no VGPR data) ----
// lds_off: LDS byte address (low 32 bits of generic shared pointer)
// goff   : 32-bit byte offset added to sgpr base (GVS addressing)
__device__ __forceinline__ void async_g2l_b128(unsigned lds_off, unsigned goff,
                                               const void* sbase) {
  asm volatile("global_load_async_to_lds_b128 %0, %1, %2"
               : : "v"(lds_off), "v"(goff), "s"(sbase) : "memory");
}
__device__ __forceinline__ void wait_async() {
  asm volatile("s_wait_asynccnt 0" : : : "memory");
}
__device__ __forceinline__ unsigned lds_addr(const void* p) {
  return (unsigned)(unsigned long long)p;  // addr[31:0] = LDS byte offset
}

// ---------- prep kernels ----------
// x (B,C,N,T) fp32 -> xb bf16 in [b][t][n][c] (channel innermost)
__global__ __launch_bounds__(256) void k_prep_x(const float* __restrict__ x,
                                                __bf16* __restrict__ xb) {
  size_t idx = (size_t)blockIdx.x * 256 + threadIdx.x;
  int t = (int)(idx & 63);
  int n = (int)((idx >> 6) & 511);
  int c = (int)((idx >> 15) & 63);
  int b = (int)(idx >> 21);
  xb[(((size_t)b * Tc + t) * Nc + n) * Cc + c] = (__bf16)x[idx];
}

// adj (K,N,N) fp32 -> adjT bf16 [k][n][m] = adj[k][m][n]
__global__ __launch_bounds__(256) void k_prep_adj(const float* __restrict__ adj,
                                                  __bf16* __restrict__ adjT) {
  size_t idx = (size_t)blockIdx.x * 256 + threadIdx.x;
  int nn = (int)(idx & 511);
  int m  = (int)((idx >> 9) & 511);
  int k  = (int)(idx >> 18);
  adjT[((size_t)k * Nc + nn) * Nc + m] = (__bf16)adj[idx];
}

// small weights -> bf16, B-operand layout [o][i] (reduction contiguous per column)
__global__ __launch_bounds__(256) void k_prep_w(
    const float* __restrict__ w1, const float* __restrict__ w2,
    const float* __restrict__ cheb, const float* __restrict__ gate_w,
    const float* __restrict__ res_w, __bf16* __restrict__ wt1,
    __bf16* __restrict__ wt2, __bf16* __restrict__ wch, __bf16* __restrict__ gw,
    __bf16* __restrict__ rw, float* __restrict__ stats) {
  int tid = threadIdx.x;
  for (int idx = tid; idx < 3 * 64 * 64; idx += 256) {
    int dt = idx >> 12, r = idx & 4095, o = r >> 6, i = r & 63;
    wt1[idx] = (__bf16)w1[((o << 6) | i) * 3 + dt];   // w1:(o,i,1,3) -> [dt][o][i]
    wt2[idx] = (__bf16)w2[((o << 6) | i) * 3 + dt];
    wch[idx] = (__bf16)cheb[(dt << 12) | (i << 6) | o]; // cheb:(k,i,o) -> [k][o][i]
  }
  for (int idx = tid; idx < 64 * 128; idx += 256) gw[idx] = (__bf16)gate_w[idx];
  for (int idx = tid; idx < 64 * 64; idx += 256)  rw[idx] = (__bf16)res_w[idx];
  for (int idx = tid; idx < 6 * 64; idx += 256)   stats[idx] = 0.f;
}

// ---------- temporal (1,3) conv as WMMA GEMM per (b,n) ----------
__global__ __launch_bounds__(256) void k_conv_t3(
    const __bf16* __restrict__ act, const __bf16* __restrict__ wt,
    const float* __restrict__ bias, float* __restrict__ raw,
    float* __restrict__ gsum, float* __restrict__ gsq) {
  __shared__ __bf16 Xs[66 * 72];       // rows t+1, pad-zero halo rows 0 & 65
  __shared__ __bf16 Ws[3 * 64 * 64];   // [dt][o][i]
  __shared__ float lsum[64], lsq[64];
  const int tid = threadIdx.x;
  const int b = blockIdx.x >> 9, n = blockIdx.x & 511;

  for (int i = tid; i < 66 * 72; i += 256) Xs[i] = (__bf16)0.f;
  if (tid < 64) { lsum[tid] = 0.f; lsq[tid] = 0.f; }
  __syncthreads();

  // async stage: weights (24KB contiguous) + 64 activation rows (128B each)
  {
    const unsigned wbase = lds_addr(Ws);
    for (int i = tid; i < 1536; i += 256)  // 1536 x 16B = 24576B
      async_g2l_b128(wbase + i * 16u, (unsigned)i * 16u, wt);
    const void* abase = act + (((size_t)b * Tc) * Nc + n) * Cc;
    const unsigned xbase = lds_addr(Xs);
    for (int i = tid; i < 512; i += 256) {  // 64 rows x 8 chunks
      int t = i >> 3, ch = i & 7;
      async_g2l_b128(xbase + (unsigned)(t + 1) * 144u + (unsigned)ch * 16u,
                     (unsigned)t * (Nc * Cc * 2u) + (unsigned)ch * 16u, abase);
    }
  }
  wait_async();
  __syncthreads();

  const int lane = tid & 31, wv = tid >> 5, hi = lane >> 4, lo = lane & 15;
  const int tr = wv >> 1, tc0 = (wv & 1) * 2;  // wave owns rows tr, cols tc0,tc0+1

  v8f acc0 = {0, 0, 0, 0, 0, 0, 0, 0};
  v8f acc1 = {0, 0, 0, 0, 0, 0, 0, 0};
  for (int dt = 0; dt < 3; ++dt)
    for (int ks = 0; ks < 2; ++ks) {
      int koff = ks * 32 + hi * 16;
      v16bf a = *(const v16bf*)&Xs[(tr * 16 + lo + dt) * 72 + koff];  // shared A
      v16bf b0 = *(const v16bf*)&Ws[dt * 4096 + (tc0 * 16 + lo) * 64 + koff];
      v16bf b1 = *(const v16bf*)&Ws[dt * 4096 + ((tc0 + 1) * 16 + lo) * 64 + koff];
      acc0 = wmma_bf16(a, b0, acc0);
      acc1 = wmma_bf16(a, b1, acc1);
    }
  for (int tl = 0; tl < 2; ++tl) {
    const v8f& acc = tl ? acc1 : acc0;
    int o = (tc0 + tl) * 16 + lo;
    float bo = bias[o], s = 0.f, s2 = 0.f;
    for (int r = 0; r < 8; ++r) {
      float v = acc[r] + bo;
      int t = tr * 16 + r + hi * 8;
      raw[(((size_t)b * Tc + t) * Nc + n) * Cc + o] = v;
      s += v; s2 += v * v;
    }
    atomicAdd(&lsum[o], s);
    atomicAdd(&lsq[o], s2);
  }
  __syncthreads();
  if (tid < 64) { atomicAdd(&gsum[tid], lsum[tid]); atomicAdd(&gsq[tid], lsq[tid]); }
}

// BN (training stats) + ReLU, fp32 -> bf16
__global__ __launch_bounds__(256) void k_bnrelu(
    const float* __restrict__ raw, const float* __restrict__ gsum,
    const float* __restrict__ gsq, const float* __restrict__ g,
    const float* __restrict__ be, __bf16* __restrict__ out) {
  size_t idx = (size_t)blockIdx.x * 256 + threadIdx.x;
  int c = (int)(idx & 63);
  float inv = 1.f / CNT;
  float mean = gsum[c] * inv;
  float var  = gsq[c] * inv - mean * mean;
  float sc = g[c] * rsqrtf(var + EPS);
  float v = (raw[idx] - mean) * sc + be[c];
  out[idx] = (__bf16)(v > 0.f ? v : 0.f);
}

// ---------- spatial Chebyshev: (adjT x) then (.. cheb_w), k-accumulated ----------
__global__ __launch_bounds__(256) void k_spatial(
    const __bf16* __restrict__ xb, const __bf16* __restrict__ adjT,
    const __bf16* __restrict__ wch, const float* __restrict__ cheb_b,
    float* __restrict__ sp) {
  __shared__ __bf16 Aj[64 * 136];  // [n_local][m_chunk]
  __shared__ __bf16 Xs[64 * 136];  // [i][m_chunk]   (B operand: column=K-contig)
  __shared__ __bf16 Tn[64 * 72];   // tmp [n_local][i] bf16
  __shared__ __bf16 Wk[64 * 64];   // cheb [o][i]
  const int tid = threadIdx.x;
  const int bt = blockIdx.x >> 3, nt = blockIdx.x & 7;
  const int lane = tid & 31, wv = tid >> 5, hi = lane >> 4, lo = lane & 15;
  const int tr = wv >> 1, tc0 = (wv & 1) * 2;

  v8f oacc0 = {0, 0, 0, 0, 0, 0, 0, 0};
  v8f oacc1 = {0, 0, 0, 0, 0, 0, 0, 0};
  for (int k = 0; k < 3; ++k) {
    __syncthreads();
    {  // async: cheb_w block for this k (8KB contiguous)
      const unsigned wkb = lds_addr(Wk);
      const __bf16* wsrc = wch + k * 4096;
      for (int i = tid; i < 512; i += 256)
        async_g2l_b128(wkb + (unsigned)i * 16u, (unsigned)i * 16u, wsrc);
    }
    v8f tacc0 = {0, 0, 0, 0, 0, 0, 0, 0};
    v8f tacc1 = {0, 0, 0, 0, 0, 0, 0, 0};
    for (int mc = 0; mc < 4; ++mc) {
      int mb = mc * 128;
      __syncthreads();
      {  // async: adj rows (64 x 256B, row-major both sides)
        const void* abase = adjT + ((size_t)k * Nc + nt * 64) * Nc + mb;
        const unsigned ajb = lds_addr(Aj);
        for (int i = tid; i < 1024; i += 256) {  // 64 rows x 16 chunks
          int r = i >> 4, ch = i & 15;
          async_g2l_b128(ajb + (unsigned)r * 272u + (unsigned)ch * 16u,
                         (unsigned)r * (Nc * 2u) + (unsigned)ch * 16u, abase);
        }
      }
      // transpose-stage x chunk manually: [m][i] global -> [i][m] LDS
      for (int i = tid; i < 128 * 64; i += 256) {
        int mm = i >> 6, ci = i & 63;
        Xs[ci * 136 + mm] = xb[((size_t)bt * Nc + mb + mm) * Cc + ci];
      }
      wait_async();
      __syncthreads();
      for (int ks = 0; ks < 4; ++ks) {
        int koff = ks * 32 + hi * 16;
        v16bf a  = *(const v16bf*)&Aj[(tr * 16 + lo) * 136 + koff];  // shared A
        v16bf b0 = *(const v16bf*)&Xs[(tc0 * 16 + lo) * 136 + koff];
        v16bf b1 = *(const v16bf*)&Xs[((tc0 + 1) * 16 + lo) * 136 + koff];
        tacc0 = wmma_bf16(a, b0, tacc0);
        tacc1 = wmma_bf16(a, b1, tacc1);
      }
    }
    __syncthreads();
    for (int r = 0; r < 8; ++r) {
      int nr = tr * 16 + r + hi * 8;
      Tn[nr * 72 + tc0 * 16 + lo]       = (__bf16)tacc0[r];
      Tn[nr * 72 + (tc0 + 1) * 16 + lo] = (__bf16)tacc1[r];
    }
    __syncthreads();
    for (int ks = 0; ks < 2; ++ks) {
      int koff = ks * 32 + hi * 16;
      v16bf a  = *(const v16bf*)&Tn[(tr * 16 + lo) * 72 + koff];  // shared A
      v16bf b0 = *(const v16bf*)&Wk[(tc0 * 16 + lo) * 64 + koff];
      v16bf b1 = *(const v16bf*)&Wk[((tc0 + 1) * 16 + lo) * 64 + koff];
      oacc0 = wmma_bf16(a, b0, oacc0);
      oacc1 = wmma_bf16(a, b1, oacc1);
    }
  }
  for (int tl = 0; tl < 2; ++tl) {
    const v8f& oacc = tl ? oacc1 : oacc0;
    int o = (tc0 + tl) * 16 + lo;
    float bo = cheb_b[o];
    for (int r = 0; r < 8; ++r) {
      int ng = nt * 64 + tr * 16 + r + hi * 8;
      sp[((size_t)bt * Nc + ng) * Cc + o] = oacc[r] + bo;
    }
  }
}

// ---------- gated fusion + residual + BN3 stats ----------
__global__ __launch_bounds__(256) void k_fuse(
    const float* __restrict__ sp, const __bf16* __restrict__ h2b,
    const __bf16* __restrict__ xb, const __bf16* __restrict__ gw,
    const __bf16* __restrict__ rw, const float* __restrict__ gate_b,
    const float* __restrict__ res_b, float* __restrict__ opre,
    float* __restrict__ gsum, float* __restrict__ gsq) {
  __shared__ __bf16 Sb[64 * 72], H2[64 * 72], Xb[64 * 72];
  __shared__ __bf16 GW[64 * 128], RW[64 * 64];
  __shared__ float lsum[64], lsq[64];
  const int tid = threadIdx.x;
  const int b = blockIdx.x >> 9, n = blockIdx.x & 511;

  {  // async: h2/x rows (contiguous 128B per t-row), gate/res weights
    const size_t rowbase = (((size_t)b * Tc) * Nc + n) * Cc;
    const void* hsrc = h2b + rowbase;
    const void* xsrc = xb + rowbase;
    const unsigned h2a = lds_addr(H2), xba = lds_addr(Xb);
    for (int i = tid; i < 512; i += 256) {  // 64 rows x 8 chunks
      int t = i >> 3, ch = i & 7;
      unsigned lo_off = (unsigned)t * 144u + (unsigned)ch * 16u;
      unsigned go = (unsigned)t * (Nc * Cc * 2u) + (unsigned)ch * 16u;
      async_g2l_b128(h2a + lo_off, go, hsrc);
      async_g2l_b128(xba + lo_off, go, xsrc);
    }
    const unsigned gwa = lds_addr(GW), rwa = lds_addr(RW);
    for (int i = tid; i < 1024; i += 256)
      async_g2l_b128(gwa + (unsigned)i * 16u, (unsigned)i * 16u, gw);
    for (int i = tid; i < 512; i += 256)
      async_g2l_b128(rwa + (unsigned)i * 16u, (unsigned)i * 16u, rw);
  }
  for (int i = tid; i < 64 * 64; i += 256) {  // spatial: fp32 -> bf16 (manual)
    int t = i >> 6, ci = i & 63;
    Sb[t * 72 + ci] = (__bf16)sp[(((size_t)b * Tc + t) * Nc + n) * Cc + ci];
  }
  if (tid < 64) { lsum[tid] = 0.f; lsq[tid] = 0.f; }
  wait_async();
  __syncthreads();

  const int lane = tid & 31, wv = tid >> 5, hi = lane >> 4, lo = lane & 15;
  const int tr = wv >> 1, tc0 = (wv & 1) * 2;

  v8f gacc0 = {0, 0, 0, 0, 0, 0, 0, 0}, gacc1 = {0, 0, 0, 0, 0, 0, 0, 0};
  v8f racc0 = {0, 0, 0, 0, 0, 0, 0, 0}, racc1 = {0, 0, 0, 0, 0, 0, 0, 0};
  for (int ks = 0; ks < 4; ++ks) {  // K=128: [spatial(64) | t(64)]
    int koff = ks * 32 + hi * 16;
    v16bf a = (koff < 64)
                  ? *(const v16bf*)&Sb[(tr * 16 + lo) * 72 + koff]
                  : *(const v16bf*)&H2[(tr * 16 + lo) * 72 + (koff - 64)];
    v16bf b0 = *(const v16bf*)&GW[(tc0 * 16 + lo) * 128 + koff];
    v16bf b1 = *(const v16bf*)&GW[((tc0 + 1) * 16 + lo) * 128 + koff];
    gacc0 = wmma_bf16(a, b0, gacc0);
    gacc1 = wmma_bf16(a, b1, gacc1);
  }
  for (int ks = 0; ks < 2; ++ks) {  // residual 1x1, K=64
    int koff = ks * 32 + hi * 16;
    v16bf a  = *(const v16bf*)&Xb[(tr * 16 + lo) * 72 + koff];
    v16bf b0 = *(const v16bf*)&RW[(tc0 * 16 + lo) * 64 + koff];
    v16bf b1 = *(const v16bf*)&RW[((tc0 + 1) * 16 + lo) * 64 + koff];
    racc0 = wmma_bf16(a, b0, racc0);
    racc1 = wmma_bf16(a, b1, racc1);
  }
  for (int tl = 0; tl < 2; ++tl) {
    const v8f& gacc = tl ? gacc1 : gacc0;
    const v8f& racc = tl ? racc1 : racc0;
    int o = (tc0 + tl) * 16 + lo;
    float gb = gate_b[o], rb = res_b[o], s = 0.f, s2 = 0.f;
    for (int r = 0; r < 8; ++r) {
      int t = tr * 16 + r + hi * 8;
      float gate = 1.f / (1.f + __expf(-(gacc[r] + gb)));
      float sv = (float)Sb[t * 72 + o];
      float tv = (float)H2[t * 72 + o];
      float v = gate * sv + (1.f - gate) * tv + racc[r] + rb;
      opre[(((size_t)b * Tc + t) * Nc + n) * Cc + o] = v;
      s += v; s2 += v * v;
    }
    atomicAdd(&lsum[o], s);
    atomicAdd(&lsq[o], s2);
  }
  __syncthreads();
  if (tid < 64) { atomicAdd(&gsum[tid], lsum[tid]); atomicAdd(&gsq[tid], lsq[tid]); }
}

// final BN3 + transpose back to (B,C,N,T)
__global__ __launch_bounds__(256) void k_bn3(
    const float* __restrict__ opre, const float* __restrict__ gsum,
    const float* __restrict__ gsq, const float* __restrict__ g3,
    const float* __restrict__ be3, float* __restrict__ out) {
  size_t idx = (size_t)blockIdx.x * 256 + threadIdx.x;
  int t = (int)(idx & 63);
  int n = (int)((idx >> 6) & 511);
  int c = (int)((idx >> 15) & 63);
  int b = (int)(idx >> 21);
  float inv = 1.f / CNT;
  float mean = gsum[c] * inv;
  float var  = gsq[c] * inv - mean * mean;
  float sc = g3[c] * rsqrtf(var + EPS);
  float v = opre[(((size_t)b * Tc + t) * Nc + n) * Cc + c];
  out[idx] = (v - mean) * sc + be3[c];
}

// ---------- host launcher ----------
extern "C" void kernel_launch(void* const* d_in, const int* in_sizes, int n_in,
                              void* d_out, int out_size, void* d_ws, size_t ws_size,
                              hipStream_t stream) {
  (void)in_sizes; (void)n_in; (void)out_size; (void)ws_size;
  const float* x      = (const float*)d_in[0];
  const float* adj    = (const float*)d_in[1];
  const float* cheb_w = (const float*)d_in[2];
  const float* cheb_b = (const float*)d_in[3];
  const float* w1     = (const float*)d_in[4];
  const float* b1     = (const float*)d_in[5];
  const float* g1     = (const float*)d_in[6];
  const float* be1    = (const float*)d_in[7];
  const float* w2     = (const float*)d_in[8];
  const float* b2     = (const float*)d_in[9];
  const float* g2     = (const float*)d_in[10];
  const float* be2    = (const float*)d_in[11];
  const float* gate_w = (const float*)d_in[12];
  const float* gate_b = (const float*)d_in[13];
  const float* res_w  = (const float*)d_in[14];
  const float* res_b  = (const float*)d_in[15];
  const float* g3     = (const float*)d_in[16];
  const float* be3    = (const float*)d_in[17];

  char* w = (char*)d_ws;
  size_t off = 0;
  auto alloc = [&](size_t bytes) {
    size_t o = off;
    off = (off + bytes + 255) & ~(size_t)255;
    return o;
  };
  __bf16* xb   = (__bf16*)(w + alloc(NE * 2));
  __bf16* adjT = (__bf16*)(w + alloc((size_t)Kc * Nc * Nc * 2));
  __bf16* wt1  = (__bf16*)(w + alloc(3 * 64 * 64 * 2));
  __bf16* wt2  = (__bf16*)(w + alloc(3 * 64 * 64 * 2));
  __bf16* wch  = (__bf16*)(w + alloc(3 * 64 * 64 * 2));
  __bf16* gwb  = (__bf16*)(w + alloc(64 * 128 * 2));
  __bf16* rwb  = (__bf16*)(w + alloc(64 * 64 * 2));
  float* stats = (float*)(w + alloc(6 * 64 * 4));
  float* raw   = (float*)(w + alloc(NE * 4));   // conv1 -> conv2 -> opre (reused)
  __bf16* h1b  = (__bf16*)(w + alloc(NE * 2));
  __bf16* h2b  = (__bf16*)(w + alloc(NE * 2));
  float* sp    = (float*)(w + alloc(NE * 4));

  float* sum1 = stats + 0,   *sq1 = stats + 64;
  float* sum2 = stats + 128, *sq2 = stats + 192;
  float* sum3 = stats + 256, *sq3 = stats + 320;

  const int gridE = (int)(NE / 256);       // 131072 elementwise blocks
  const int gridBN = Bc * Nc;              // 8192 (b,n) GEMM blocks
  const int gridBT = Bc * Tc * (Nc / 64);  // 8192 (b,t,n-tile) GEMM blocks

  k_prep_w<<<1, 256, 0, stream>>>(w1, w2, cheb_w, gate_w, res_w, wt1, wt2, wch,
                                  gwb, rwb, stats);
  k_prep_x<<<gridE, 256, 0, stream>>>(x, xb);
  k_prep_adj<<<(Kc * Nc * Nc) / 256, 256, 0, stream>>>(adj, adjT);

  // temporal branch
  k_conv_t3<<<gridBN, 256, 0, stream>>>(xb, wt1, b1, raw, sum1, sq1);
  k_bnrelu<<<gridE, 256, 0, stream>>>(raw, sum1, sq1, g1, be1, h1b);
  k_conv_t3<<<gridBN, 256, 0, stream>>>(h1b, wt2, b2, raw, sum2, sq2);
  k_bnrelu<<<gridE, 256, 0, stream>>>(raw, sum2, sq2, g2, be2, h2b);

  // spatial branch
  k_spatial<<<gridBT, 256, 0, stream>>>(xb, adjT, wch, cheb_b, sp);

  // gated fusion (+ BN3 stats) then final BN + transpose
  k_fuse<<<gridBN, 256, 0, stream>>>(sp, h2b, xb, gwb, rwb, gate_b, res_b, raw,
                                     sum3, sq3);
  k_bn3<<<gridE, 256, 0, stream>>>(raw, sum3, sq3, g3, be3, (float*)d_out);
}
